// SwinTransformerBlock_45981919871212
// MI455X (gfx1250) — compile-verified, matrix operand order checked by reading, not run on hardware
//
#include <hip/hip_runtime.h>
#include <hip/hip_bf16.h>

typedef __bf16 bf16_t;
typedef __attribute__((ext_vector_type(16))) __bf16 v16bf;
typedef __attribute__((ext_vector_type(8)))  __bf16 v8bf;
typedef __attribute__((ext_vector_type(4)))  __bf16 v4bf;
typedef __attribute__((ext_vector_type(8)))  float  v8f;
typedef __attribute__((ext_vector_type(4)))  float  v4f;
typedef int v4i_vs __attribute__((vector_size(16)));   // matches async-LDS builtin param

// Problem constants (SwinV2 block)
constexpr int  CB    = 32;              // batch
constexpr int  CH    = 56;              // height/width
constexpr int  CC    = 512;             // channels
constexpr int  NH    = 16;              // heads
constexpr int  NTOK  = 49;              // tokens per window (7x7)
constexpr int  NWTOT = CB * 64;         // 2048 windows
constexpr long MTOK  = (long)NWTOT * NTOK; // 100352 token rows

// Async global->LDS (CDNA5). Guarded so compilation never regresses.
#if defined(__has_builtin)
#if __has_builtin(__builtin_amdgcn_global_load_async_to_lds_b128)
#define HAVE_ASYNC_LDS 1
#endif
#endif
#ifndef HAVE_ASYNC_LDS
#define HAVE_ASYNC_LDS 0
#endif

__device__ inline void wait_async_lds() {
#if HAVE_ASYNC_LDS
#if defined(__has_builtin) && __has_builtin(__builtin_amdgcn_s_wait_asynccnt)
  __builtin_amdgcn_s_wait_asynccnt(0);
#else
  asm volatile("s_wait_asynccnt 0" ::: "memory");
#endif
#endif
}

// ---------------------------------------------------------------------------
// WMMA helpers (gfx1250, wave32). bf16 16x16x32, f32 accumulate.
// A fragment per lane: row = lane&15; k chunks [8*(lane>>4), +8) and +16.
// B fragment per lane: col = lane&15; same k chunking (weights row-major in K).
// C fragment element (reg r, lane l): row = r + 8*(l>>4), col = l&15.
// ---------------------------------------------------------------------------
__device__ inline v16bf load_frag16(const bf16_t* p) {
  v8bf lo = *(const v8bf*)(p);
  v8bf hi = *(const v8bf*)(p + 16);
  return __builtin_shufflevector(lo, hi, 0,1,2,3,4,5,6,7,8,9,10,11,12,13,14,15);
}
__device__ inline v8f wmma_bf16(v16bf a, v16bf b, v8f c) {
  return __builtin_amdgcn_wmma_f32_16x16x32_bf16(false, a, false, b, (short)0, c,
                                                 false, false);
}

// ---------------------------------------------------------------------------
// Generic f32 -> bf16 convert (weights)
// ---------------------------------------------------------------------------
__global__ __launch_bounds__(256) void cvt_bf16_kernel(const float* __restrict__ src,
                                                       bf16_t* __restrict__ dst, long n) {
  long i = ((long)blockIdx.x * 256 + threadIdx.x) * 4;
  if (i >= n) return;
  v4f v = *(const v4f*)(src + i);
  v4bf o;
  o[0] = (bf16_t)v[0]; o[1] = (bf16_t)v[1]; o[2] = (bf16_t)v[2]; o[3] = (bf16_t)v[3];
  *(v4bf*)(dst + i) = o;
}

// ---------------------------------------------------------------------------
// Fused roll(-3,-3) + window partition + bf16 convert: x (B,HW,C) -> xw (M,512)
// ---------------------------------------------------------------------------
__global__ __launch_bounds__(256) void xprep_kernel(const float* __restrict__ x,
                                                    bf16_t* __restrict__ xw) {
  long idx = ((long)blockIdx.x * 256 + threadIdx.x) * 4;
  if (idx >= MTOK * 512) return;
  long t = idx >> 9; int c = (int)(idx & 511);
  long b_ = t / NTOK; int nn = (int)(t % NTOK);
  int b = (int)(b_ >> 6), wi = (int)(b_ & 63);
  int wh = wi >> 3, ww = wi & 7;
  int i = nn / 7, j = nn % 7;
  int hs = wh * 7 + i + 3; if (hs >= CH) hs -= CH;   // roll source coord
  int wsrc = ww * 7 + j + 3; if (wsrc >= CH) wsrc -= CH;
  const float* src = x + (((long)b * (CH * CH)) + hs * CH + wsrc) * CC + c;
  v4f v = *(const v4f*)src;
  v4bf o;
  o[0] = (bf16_t)v[0]; o[1] = (bf16_t)v[1]; o[2] = (bf16_t)v[2]; o[3] = (bf16_t)v[3];
  *(v4bf*)(xw + idx) = o;
}

// ---------------------------------------------------------------------------
// CPB MLP: bias_tbl[169][16] = relu(tbl @ W1^T + b1) @ W2^T
// ---------------------------------------------------------------------------
__device__ inline float cpb_coord(int rel) {
  float v = (float)(rel - 6) * (8.0f / 6.0f);
  return copysignf(log2f(fabsf(v) + 1.0f) * (1.0f / 3.0f), v);
}
__global__ __launch_bounds__(512) void cpb_hidden_kernel(const float* __restrict__ w1,
                                                         const float* __restrict__ b1,
                                                         const float* __restrict__ w2,
                                                         float* __restrict__ bias_tbl) {
  __shared__ float hbuf[512];
  int r = blockIdx.x;
  float t0 = cpb_coord(r / 13);
  float t1 = cpb_coord(r % 13);
  int j = threadIdx.x;
  hbuf[j] = fmaxf(0.f, t0 * w1[j * 2] + t1 * w1[j * 2 + 1] + b1[j]);
  __syncthreads();
  if (j < NH) {
    float s = 0.f;
    for (int q = 0; q < 512; q++) s += hbuf[q] * w2[j * 512 + q];
    bias_tbl[r * NH + j] = s;
  }
}

// rpb[h][n][m] = 16*sigmoid(bias_tbl[rpi(n,m)][h])
__global__ __launch_bounds__(256) void rpb_kernel(const float* __restrict__ bias_tbl,
                                                  float* __restrict__ rpb) {
  int idx = blockIdx.x * 256 + threadIdx.x;
  if (idx >= NH * NTOK * NTOK) return;
  int h = idx / (NTOK * NTOK), rem = idx % (NTOK * NTOK);
  int n = rem / NTOK, m = rem % NTOK;
  int r = ((n / 7) - (m / 7) + 6) * 13 + ((n % 7) - (m % 7) + 6);
  float v = bias_tbl[r * NH + h];
  rpb[idx] = 16.f / (1.f + __expf(-v));
}

// ---------------------------------------------------------------------------
// Shared WMMA GEMM: out[m][n] = sum_k A[m][k] * W[n][k] (+epilogue)
// Block = 128 threads (4 waves), 64x64 tile, wave = 32x32 (2x2 WMMA).
// K pipelined through LDS with double-buffered async global->LDS copies.
// ---------------------------------------------------------------------------
enum { EPI_QKV = 0, EPI_PROJ = 1, EPI_GELU = 2, EPI_BIASF = 3 };

constexpr int LDS_PITCH = 40;  // bf16 elems per row (80B: 16B-aligned, bank-spread)

struct GemmArgs {
  const bf16_t* A; const bf16_t* W;
  int K; int Nld;
  const float* bias0; const float* bias1;
  bf16_t* oq; bf16_t* ok; bf16_t* ovt;   // QKV epilogue
  float*  of;                             // PROJ / BIASF f32 out
  bf16_t* ob;                             // GELU bf16 out
};

// Cooperatively stage one 64x32 bf16 tile into LDS (256 x 16B transfers).
__device__ inline void stage_tile(const bf16_t* __restrict__ gbase, long row0,
                                  int K, int k0, bf16_t* lds) {
  const int tid = threadIdx.x;
#pragma unroll
  for (int it = 0; it < 2; ++it) {
    int t2 = tid + 128 * it;
    int r = t2 >> 2, seg = t2 & 3;
    const bf16_t* gp = gbase + (row0 + r) * (long)K + k0 + seg * 8;
    bf16_t* lp = lds + r * LDS_PITCH + seg * 8;
#if HAVE_ASYNC_LDS
    __builtin_amdgcn_global_load_async_to_lds_b128((v4i_vs*)gp, (v4i_vs*)lp, 0, 0);
#else
    *(v8bf*)lp = *(const v8bf*)gp;
#endif
  }
}

template<int EPI>
__global__ __launch_bounds__(128) void gemm_wmma_kernel(GemmArgs g) {
  __shared__ bf16_t Abuf[2][64 * LDS_PITCH];
  __shared__ bf16_t Bbuf[2][64 * LDS_PITCH];

  const int lane = threadIdx.x & 31;
  const int wid  = threadIdx.x >> 5;
  const int lrow = lane & 15, lhi = lane >> 4;
  const long m0 = (long)blockIdx.y * 64 + (wid >> 1) * 32;
  const long n0 = (long)blockIdx.x * 64 + (wid & 1) * 32;
  const long mblk = (long)blockIdx.y * 64;
  const long nblk = (long)blockIdx.x * 64;
  const int K = g.K;
  const int S = K >> 5;

  v8f acc[2][2] = {};

  stage_tile(g.A, mblk, K, 0, Abuf[0]);
  stage_tile(g.W, nblk, K, 0, Bbuf[0]);

  const int arow = (wid >> 1) * 32 + lrow;   // local row in A tile (+16 for rt=1)
  const int brow = (wid & 1) * 32 + lrow;    // local row in B tile (+16 for ct=1)

  for (int s = 0; s < S; ++s) {
    const int cur = s & 1;
    wait_async_lds();          // buf[cur] landed (issued last iteration / prologue)
    __syncthreads();           // visible to all waves; all past reads of buf[1-cur]
    if (s + 1 < S) {
      stage_tile(g.A, mblk, K, (s + 1) * 32, Abuf[1 - cur]);
      stage_tile(g.W, nblk, K, (s + 1) * 32, Bbuf[1 - cur]);
    }
    const bf16_t* Al = Abuf[cur];
    const bf16_t* Bl = Bbuf[cur];
    v16bf a0 = load_frag16(Al + arow * LDS_PITCH + 8 * lhi);
    v16bf a1 = load_frag16(Al + (arow + 16) * LDS_PITCH + 8 * lhi);
    v16bf b0 = load_frag16(Bl + brow * LDS_PITCH + 8 * lhi);
    v16bf b1 = load_frag16(Bl + (brow + 16) * LDS_PITCH + 8 * lhi);
    acc[0][0] = wmma_bf16(a0, b0, acc[0][0]);
    acc[0][1] = wmma_bf16(a0, b1, acc[0][1]);
    acc[1][0] = wmma_bf16(a1, b0, acc[1][0]);
    acc[1][1] = wmma_bf16(a1, b1, acc[1][1]);
  }

  const int rbase = 8 * lhi;

  if (EPI == EPI_QKV) {
    // wave's 32 cols = exactly one head within one of q/k/v sections
    const int sec  = (int)(n0 >> 9);
    const int c512 = (int)(n0 & 511);
    const int head = c512 >> 5;
    float bias[2];
#pragma unroll
    for (int ct = 0; ct < 2; ct++) {
      int c = c512 + ct * 16 + lrow;
      bias[ct] = (sec == 0) ? g.bias0[c] : (sec == 2 ? g.bias1[c] : 0.f);
    }
#pragma unroll
    for (int rt = 0; rt < 2; rt++) {
#pragma unroll
      for (int r = 0; r < 8; r++) {
        float v0 = acc[rt][0][r] + bias[0];
        float v1 = acc[rt][1][r] + bias[1];
        if (sec < 2) {  // cosine-attention L2 normalization over head dim (32)
          float ss = v0 * v0 + v1 * v1;
#pragma unroll
          for (int o = 1; o < 16; o <<= 1) ss += __shfl_xor(ss, o, 16);
          float inv = 1.f / fmaxf(sqrtf(ss), 1e-12f);
          v0 *= inv; v1 *= inv;
        }
        long m = m0 + rt * 16 + r + rbase;
        long b_ = m / NTOK; int nn = (int)(m % NTOK);
        if (sec == 2) {  // store v transposed: vt[(b_,h,d)][n]
          long base = (b_ * NH + head) * 32;
          g.ovt[(base + lrow) * 64 + nn]      = (bf16_t)v0;
          g.ovt[(base + 16 + lrow) * 64 + nn] = (bf16_t)v1;
        } else {
          bf16_t* dst = (sec == 0) ? g.oq : g.ok;
          long rowoff = m * 512 + c512;
          dst[rowoff + lrow]      = (bf16_t)v0;
          dst[rowoff + 16 + lrow] = (bf16_t)v1;
        }
      }
    }
  } else if (EPI == EPI_PROJ) {
#pragma unroll
    for (int rt = 0; rt < 2; rt++)
#pragma unroll
      for (int ct = 0; ct < 2; ct++) {
        int c = (int)n0 + ct * 16 + lrow;
        float bias = g.bias0[c];
#pragma unroll
        for (int r = 0; r < 8; r++) {
          long m = m0 + rt * 16 + r + rbase;
          long b_ = m / NTOK; int nn = (int)(m % NTOK);
          int b = (int)(b_ >> 6), wi = (int)(b_ & 63);
          int wh = wi >> 3, ww = wi & 7;
          int i = nn / 7, j = nn % 7;
          int hh = wh * 7 + i + 3; if (hh >= CH) hh -= CH;   // reverse roll
          int wc = ww * 7 + j + 3; if (wc >= CH) wc -= CH;
          g.of[(((long)b * (CH * CH)) + hh * CH + wc) * 512 + c] = acc[rt][ct][r] + bias;
        }
      }
  } else {
#pragma unroll
    for (int rt = 0; rt < 2; rt++)
#pragma unroll
      for (int ct = 0; ct < 2; ct++) {
        int c = (int)n0 + ct * 16 + lrow;
        float bias = g.bias0[c];
#pragma unroll
        for (int r = 0; r < 8; r++) {
          long m = m0 + rt * 16 + r + rbase;
          float v = acc[rt][ct][r] + bias;
          if (EPI == EPI_GELU) {
            float ge = 0.5f * v * (1.f + erff(v * 0.70710678118654752f));
            g.ob[m * (long)g.Nld + c] = (bf16_t)ge;
          } else {
            g.of[m * (long)g.Nld + c] = v;
          }
        }
      }
  }
}

// ---------------------------------------------------------------------------
// Attention: one wave per (window b_, head h). S = qn*kn^T (49->64 pad, K=32),
// scale + CPB bias + shift mask + softmax, O = P @ v (K=64, two WMMA steps).
// ---------------------------------------------------------------------------
__global__ __launch_bounds__(32)
void attn_kernel(const bf16_t* __restrict__ qn, const bf16_t* __restrict__ kn,
                 const bf16_t* __restrict__ vt, const float* __restrict__ rpb,
                 const float* __restrict__ logit_scale, bf16_t* __restrict__ ctx) {
  const int b_ = blockIdx.x;
  const int h  = blockIdx.y;
  const int lane = threadIdx.x;
  const int lrow = lane & 15, lhi = lane >> 4;
  __shared__ float P[64][68];

  const int wi = b_ & 63, wh = wi >> 3, ww = wi & 7;

  v16bf qf[4], kf[4];
#pragma unroll
  for (int t = 0; t < 4; t++) {
    int n = t * 16 + lrow;
    v16bf zq = {}, zk = {};
    if (n < NTOK) {
      long row = (long)(b_ * NTOK + n) * 512 + h * 32 + 8 * lhi;
      zq = load_frag16(qn + row);
      zk = load_frag16(kn + row);
    }
    qf[t] = zq; kf[t] = zk;
  }

  v8f S[4][4] = {};
#pragma unroll
  for (int mt = 0; mt < 4; mt++)
#pragma unroll
    for (int ct = 0; ct < 4; ct++)
      S[mt][ct] = wmma_bf16(qf[mt], kf[ct], S[mt][ct]);

  const float scale = __expf(fminf(logit_scale[h], 4.60517019f)); // exp(min(ls, ln100))
  const float* rb = rpb + h * (NTOK * NTOK);

  auto rid = [&](int n) -> int {
    int i = n / 7, j = n % 7;
    int hr = wh * 7 + i, wr = ww * 7 + j;
    int a = (hr < 49) ? 0 : ((hr < 53) ? 1 : 2);
    int bb = (wr < 49) ? 0 : ((wr < 53) ? 1 : 2);
    return a * 3 + bb;
  };

#pragma unroll
  for (int mt = 0; mt < 4; mt++) {
#pragma unroll
    for (int r = 0; r < 8; r++) {
      int row = mt * 16 + r + 8 * lhi;
      bool rv = row < NTOK;
      int idr = rv ? rid(row) : 0;
      float vals[4];
#pragma unroll
      for (int ct = 0; ct < 4; ct++) {
        int col = ct * 16 + lrow;
        float s = -1e30f;
        if (rv && col < NTOK) {
          float mask = (idr != rid(col)) ? -100.f : 0.f;
          s = S[mt][ct][r] * scale + rb[row * NTOK + col] + mask;
        }
        vals[ct] = s;
      }
      float mx = fmaxf(fmaxf(vals[0], vals[1]), fmaxf(vals[2], vals[3]));
#pragma unroll
      for (int o = 1; o < 16; o <<= 1) mx = fmaxf(mx, __shfl_xor(mx, o, 16));
      float sum = 0.f;
#pragma unroll
      for (int ct = 0; ct < 4; ct++) { vals[ct] = __expf(vals[ct] - mx); sum += vals[ct]; }
#pragma unroll
      for (int o = 1; o < 16; o <<= 1) sum += __shfl_xor(sum, o, 16);
      float inv = 1.f / sum;
#pragma unroll
      for (int ct = 0; ct < 4; ct++) P[row][ct * 16 + lrow] = vals[ct] * inv;
    }
  }
  __syncthreads();

  // Re-fragment P (C-layout f32 in LDS) as bf16 A-fragments
  v16bf pa[4][2];
#pragma unroll
  for (int mt = 0; mt < 4; mt++) {
    int row = mt * 16 + lrow;
#pragma unroll
    for (int ks = 0; ks < 2; ks++) {
      int kb = ks * 32 + 8 * lhi;
      v16bf f = {};
#pragma unroll
      for (int q = 0; q < 8; q++) {
        f[q]     = (bf16_t)P[row][kb + q];
        f[8 + q] = (bf16_t)P[row][kb + 16 + q];
      }
      pa[mt][ks] = f;
    }
  }
  v16bf vb[2][2];
#pragma unroll
  for (int ctd = 0; ctd < 2; ctd++) {
    int d = ctd * 16 + lrow;
    const bf16_t* pv = vt + ((long)(b_ * NH + h) * 32 + d) * 64;
#pragma unroll
    for (int ks = 0; ks < 2; ks++)
      vb[ks][ctd] = load_frag16(pv + ks * 32 + 8 * lhi);
  }
  v8f O[4][2] = {};
#pragma unroll
  for (int mt = 0; mt < 4; mt++)
#pragma unroll
    for (int ctd = 0; ctd < 2; ctd++) {
      O[mt][ctd] = wmma_bf16(pa[mt][0], vb[0][ctd], O[mt][ctd]);
      O[mt][ctd] = wmma_bf16(pa[mt][1], vb[1][ctd], O[mt][ctd]);
    }
#pragma unroll
  for (int mt = 0; mt < 4; mt++)
#pragma unroll
    for (int r = 0; r < 8; r++) {
      int row = mt * 16 + r + 8 * lhi;
      if (row < NTOK) {
        long t = (long)b_ * NTOK + row;
#pragma unroll
        for (int ctd = 0; ctd < 2; ctd++)
          ctx[t * 512 + h * 32 + ctd * 16 + lrow] = (bf16_t)O[mt][ctd][r];
      }
    }
}

// ---------------------------------------------------------------------------
// Residual + post-LayerNorm: out = base + LN(src)*g + b  (one wave per token)
// ---------------------------------------------------------------------------
template<bool WRITE_BF16>
__global__ __launch_bounds__(256)
void ln_res_kernel(const float* __restrict__ src, const float* __restrict__ base,
                   const float* __restrict__ gamma, const float* __restrict__ beta,
                   float* __restrict__ outf, bf16_t* __restrict__ outb) {
  const int lane = threadIdx.x & 31;
  const int wv = threadIdx.x >> 5;
  const long tok = (long)blockIdx.x * 8 + wv;
  const float* p = src + tok * 512;
  v4f v[4];
#pragma unroll
  for (int q = 0; q < 4; q++) v[q] = *(const v4f*)(p + q * 128 + lane * 4);
  float s = 0.f, s2 = 0.f;
#pragma unroll
  for (int q = 0; q < 4; q++)
#pragma unroll
    for (int e = 0; e < 4; e++) { float t = v[q][e]; s += t; s2 += t * t; }
#pragma unroll
  for (int o = 1; o < 32; o <<= 1) { s += __shfl_xor(s, o); s2 += __shfl_xor(s2, o); }
  float mean = s * (1.f / 512.f);
  float var  = s2 * (1.f / 512.f) - mean * mean;
  float rstd = rsqrtf(var + 1e-5f);
#pragma unroll
  for (int q = 0; q < 4; q++)
#pragma unroll
    for (int e = 0; e < 4; e++) {
      int c = q * 128 + lane * 4 + e;
      float y = (v[q][e] - mean) * rstd * gamma[c] + beta[c];
      float o = base[tok * 512 + c] + y;
      outf[tok * 512 + c] = o;
      if (WRITE_BF16) outb[tok * 512 + c] = (bf16_t)o;
    }
}

// ---------------------------------------------------------------------------
// Host launch
// ---------------------------------------------------------------------------
extern "C" void kernel_launch(void* const* d_in, const int* in_sizes, int n_in,
                              void* d_out, int out_size, void* d_ws, size_t ws_size,
                              hipStream_t stream) {
  (void)in_sizes; (void)n_in; (void)out_size;
  const float* x         = (const float*)d_in[0];
  const float* qkv_w     = (const float*)d_in[1];
  const float* q_bias    = (const float*)d_in[2];
  const float* v_bias    = (const float*)d_in[3];
  const float* logit_sc  = (const float*)d_in[4];
  const float* cpb_w1    = (const float*)d_in[5];
  const float* cpb_b1    = (const float*)d_in[6];
  const float* cpb_w2    = (const float*)d_in[7];
  const float* proj_w    = (const float*)d_in[8];
  const float* proj_b    = (const float*)d_in[9];
  const float* n1g       = (const float*)d_in[10];
  const float* n1b       = (const float*)d_in[11];
  const float* fc1_w     = (const float*)d_in[12];
  const float* fc1_b     = (const float*)d_in[13];
  const float* fc2_w     = (const float*)d_in[14];
  const float* fc2_b     = (const float*)d_in[15];
  const float* n2g       = (const float*)d_in[16];
  const float* n2b       = (const float*)d_in[17];
  float* out = (float*)d_out;

  char* ws = (char*)d_ws;
  size_t off = 0;
  auto alloc = [&](size_t bytes) -> char* {
    off = (off + 255) & ~(size_t)255;
    char* p = ws + off; off += bytes; return p;
  };
  bf16_t* wq   = (bf16_t*)alloc(1536L * 512 * 2);
  bf16_t* wp   = (bf16_t*)alloc(512L * 512 * 2);
  bf16_t* wf1  = (bf16_t*)alloc(2048L * 512 * 2);
  bf16_t* wf2  = (bf16_t*)alloc(512L * 2048 * 2);
  float*  btbl = (float*)alloc(169L * NH * 4);
  float*  rpb  = (float*)alloc((long)NH * NTOK * NTOK * 4);
  bf16_t* xw   = (bf16_t*)alloc(MTOK * 512 * 2);   // later reused as ctx
  bf16_t* qn   = (bf16_t*)alloc(MTOK * 512 * 2);
  bf16_t* kb   = (bf16_t*)alloc(MTOK * 512 * 2);
  size_t  vtsz = (size_t)NWTOT * NH * 32 * 64 * 2;
  bf16_t* vt   = (bf16_t*)alloc(vtsz);
  float*  ybuf = (float*)alloc(MTOK * 512 * 4);    // later reused as m
  float*  x1   = (float*)alloc(MTOK * 512 * 4);
  bf16_t* x1b  = (bf16_t*)alloc(MTOK * 512 * 2);
  bf16_t* h1   = (bf16_t*)alloc(MTOK * 2048 * 2);
  if (off > ws_size) return;  // insufficient scratch

  bf16_t* ctx = xw;    // alias: xw dead after QKV GEMM
  float*  m   = ybuf;  // alias: y dead after LN1

  // 1) weights -> bf16
  cvt_bf16_kernel<<<768, 256, 0, stream>>>(qkv_w, wq, 1536L * 512);
  cvt_bf16_kernel<<<256, 256, 0, stream>>>(proj_w, wp, 512L * 512);
  cvt_bf16_kernel<<<1024, 256, 0, stream>>>(fc1_w, wf1, 2048L * 512);
  cvt_bf16_kernel<<<1024, 256, 0, stream>>>(fc2_w, wf2, 512L * 2048);

  // 2) CPB MLP -> relative position bias table
  cpb_hidden_kernel<<<169, 512, 0, stream>>>(cpb_w1, cpb_b1, cpb_w2, btbl);
  rpb_kernel<<<(NH * NTOK * NTOK + 255) / 256, 256, 0, stream>>>(btbl, rpb);

  // 3) roll + window partition + bf16
  xprep_kernel<<<(int)(MTOK * 512 / 4 / 256), 256, 0, stream>>>(x, xw);

  // 4) QKV GEMM with fused bias + q/k L2-norm, v transposed (zero vt padding first)
  (void)hipMemsetAsync(vt, 0, vtsz, stream);
  {
    GemmArgs g{xw, wq, 512, 1536, q_bias, v_bias, qn, kb, vt, nullptr, nullptr};
    gemm_wmma_kernel<EPI_QKV><<<dim3(1536 / 64, (int)(MTOK / 64)), 128, 0, stream>>>(g);
  }

  // 5) attention
  attn_kernel<<<dim3(NWTOT, NH), 32, 0, stream>>>(qn, kb, vt, rpb, logit_sc, ctx);

  // 6) proj GEMM with fused reverse-roll scatter
  {
    GemmArgs g{ctx, wp, 512, 512, proj_b, nullptr, nullptr, nullptr, nullptr, ybuf, nullptr};
    gemm_wmma_kernel<EPI_PROJ><<<dim3(512 / 64, (int)(MTOK / 64)), 128, 0, stream>>>(g);
  }

  // 7) x1 = x + LN(y); also bf16 copy for fc1
  ln_res_kernel<true><<<(int)(MTOK / 8), 256, 0, stream>>>(ybuf, x, n1g, n1b, x1, x1b);

  // 8) fc1 + GELU
  {
    GemmArgs g{x1b, wf1, 512, 2048, fc1_b, nullptr, nullptr, nullptr, nullptr, nullptr, h1};
    gemm_wmma_kernel<EPI_GELU><<<dim3(2048 / 64, (int)(MTOK / 64)), 128, 0, stream>>>(g);
  }

  // 9) fc2
  {
    GemmArgs g{h1, wf2, 2048, 512, fc2_b, nullptr, nullptr, nullptr, nullptr, m, nullptr};
    gemm_wmma_kernel<EPI_BIASF><<<dim3(512 / 64, (int)(MTOK / 64)), 128, 0, stream>>>(g);
  }

  // 10) out = x1 + LN(m)
  ln_res_kernel<false><<<(int)(MTOK / 8), 256, 0, stream>>>(m, x1, n2g, n2b, out, nullptr);
}
